// Model1_7335804142294
// MI455X (gfx1250) — compile-verified, matrix operand order checked by reading, not run on hardware
//
#include <hip/hip_runtime.h>
#include <math.h>
#include <stdint.h>

// ---------------------------------------------------------------------------
// Types for CDNA5 WMMA
// ---------------------------------------------------------------------------
typedef __attribute__((ext_vector_type(16))) __bf16 v16bf;
typedef __attribute__((ext_vector_type(8)))  float  v8f;

#define BM 128
#define BN 64
#define BK 32
#define APAD 16   // LDS row pad -> row stride 48 bf16 = 96 B (16B aligned)

__device__ __forceinline__ float silu_f(float x) { return x / (1.0f + __expf(-x)); }
__device__ __forceinline__ float softplus_f(float x) { return (x > 20.0f) ? x : log1pf(__expf(x)); }

__device__ __forceinline__ unsigned pk_bf16(float a, float b)
{
    union { __bf16 h[2]; unsigned u; } p;
    p.h[0] = (__bf16)a;
    p.h[1] = (__bf16)b;
    return p.u;
}

// ---------------------------------------------------------------------------
// WMMA micro-kernel body: 8 waves, each 32x32 (2x2 tiles of 16x16)
// ---------------------------------------------------------------------------
template<int ROWLEN>
__device__ __forceinline__ void wmma_step(const __bf16 (*As)[ROWLEN],
                                          const __bf16 (*Bs)[ROWLEN],
                                          int lane, int wm, int wn,
                                          v8f acc[2][2])
{
    const int fr = lane & 15;
    const int kk = (lane >> 4) * 8;
    v16bf afrag[2], bfrag[2];
    #pragma unroll
    for (int t = 0; t < 2; ++t) {
        const int ar = wm + t * 16 + fr;
        const int br = wn + t * 16 + fr;
        #pragma unroll
        for (int i = 0; i < 4; ++i) {
            afrag[t][2 * i]         = As[ar][kk + 2 * i];
            afrag[t][2 * i + 1]     = As[ar][kk + 2 * i + 1];
            afrag[t][8 + 2 * i]     = As[ar][16 + kk + 2 * i];
            afrag[t][8 + 2 * i + 1] = As[ar][16 + kk + 2 * i + 1];
            bfrag[t][2 * i]         = Bs[br][kk + 2 * i];
            bfrag[t][2 * i + 1]     = Bs[br][kk + 2 * i + 1];
            bfrag[t][8 + 2 * i]     = Bs[br][16 + kk + 2 * i];
            bfrag[t][8 + 2 * i + 1] = Bs[br][16 + kk + 2 * i + 1];
        }
    }
    #pragma unroll
    for (int ti = 0; ti < 2; ++ti)
        #pragma unroll
        for (int tj = 0; tj < 2; ++tj)
            acc[ti][tj] = __builtin_amdgcn_wmma_f32_16x16x32_bf16(
                false, afrag[ti], false, bfrag[tj],
                (short)0, acc[ti][tj], false, false);
}

// ---------------------------------------------------------------------------
// shared epilogue: bias hoisted per column, pointer-bump stores
// (preconditions: gm always in-bounds; caller guards gn if needed)
// ---------------------------------------------------------------------------
template<int ACT, bool HAS_BIAS, bool HAS_RES>
__device__ __forceinline__ void store_tile(float* __restrict__ C, int ldc,
                                           const float* __restrict__ bias,
                                           const float* __restrict__ R, int ldr,
                                           int m0, int n0, int wm, int wn,
                                           int lane, float scale,
                                           const v8f acc[2][2])
{
    const int nn = lane & 15;
    const int mb = (lane >> 4) * 8;
    #pragma unroll
    for (int tj = 0; tj < 2; ++tj) {
        const int gn = n0 + wn + tj * 16 + nn;
        const float bv = HAS_BIAS ? bias[gn] : 0.0f;
        #pragma unroll
        for (int ti = 0; ti < 2; ++ti) {
            const int gm0 = m0 + wm + ti * 16 + mb;
            float* cp = C + gm0 * ldc + gn;
            const float* rp = HAS_RES ? (R + gm0 * ldr + gn) : nullptr;
            #pragma unroll
            for (int r = 0; r < 8; ++r) {
                float v = acc[ti][tj][r] * scale + bv;
                if (HAS_RES) { v += *rp; rp += ldr; }
                if (ACT == 1) v = silu_f(v);
                else if (ACT == 2) v = softplus_f(v);
                *cp = v;
                cp += ldc;
            }
        }
    }
}

// ---------------------------------------------------------------------------
// FAST GEMM: C = act(scale*A@B^T + bias + R).  B is (N,K) row-major.
// Preconditions: M%128==0, N%64==0, K%32==0, lda/ldb%4==0, A/B 16B-aligned.
// ---------------------------------------------------------------------------
template<int ACT, bool HAS_BIAS, bool HAS_RES>
__global__ void __launch_bounds__(256)
gemm_fast(const float* __restrict__ A, int lda, long long sA,
          const float* __restrict__ B, int ldb, long long sB,
          float* __restrict__ C, int ldc, long long sC,
          const float* __restrict__ bias,
          const float* __restrict__ R, int ldr, long long sR,
          int K, float scale)
{
    __shared__ __bf16 As[BM][BK + APAD];
    __shared__ __bf16 Bs[BN][BK + APAD];

    const int z = blockIdx.z;
    A += (long long)z * sA;
    B += (long long)z * sB;
    C += (long long)z * sC;
    if (HAS_RES) R += (long long)z * sR;

    const int m0 = blockIdx.y * BM;
    const int n0 = blockIdx.x * BN;
    const int tid  = threadIdx.x;
    const int lane = tid & 31;
    const int wave = tid >> 5;
    const int wm = (wave & 3) << 5;
    const int wn = (wave >> 2) << 5;

    const int ar = tid >> 1;               // A row 0..127
    const int ac = (tid & 1) << 4;         // 0 or 16
    const int br = tid >> 2;               // B row 0..63
    const int bc = (tid & 3) << 3;         // 0,8,16,24
    const float* asrc = A + (m0 + ar) * lda + ac;
    const float* bsrc = B + (n0 + br) * ldb + bc;
    unsigned* adst = (unsigned*)&As[ar][ac];
    unsigned* bdst = (unsigned*)&Bs[br][bc];

    v8f acc[2][2] = {};

    const int kTiles = K >> 5;
    for (int kt = 0; kt < kTiles; ++kt) {
        const float4 f0 = ((const float4*)asrc)[0];
        const float4 f1 = ((const float4*)asrc)[1];
        const float4 f2 = ((const float4*)asrc)[2];
        const float4 f3 = ((const float4*)asrc)[3];
        const float4 g0 = ((const float4*)bsrc)[0];
        const float4 g1 = ((const float4*)bsrc)[1];
        asrc += BK;
        bsrc += BK;
        adst[0] = pk_bf16(f0.x, f0.y);
        adst[1] = pk_bf16(f0.z, f0.w);
        adst[2] = pk_bf16(f1.x, f1.y);
        adst[3] = pk_bf16(f1.z, f1.w);
        adst[4] = pk_bf16(f2.x, f2.y);
        adst[5] = pk_bf16(f2.z, f2.w);
        adst[6] = pk_bf16(f3.x, f3.y);
        adst[7] = pk_bf16(f3.z, f3.w);
        bdst[0] = pk_bf16(g0.x, g0.y);
        bdst[1] = pk_bf16(g0.z, g0.w);
        bdst[2] = pk_bf16(g1.x, g1.y);
        bdst[3] = pk_bf16(g1.z, g1.w);
        __syncthreads();
        wmma_step<BK + APAD>(As, Bs, lane, wm, wn, acc);
        __syncthreads();
    }

    store_tile<ACT, HAS_BIAS, HAS_RES>(C, ldc, bias, R, ldr,
                                       m0, n0, wm, wn, lane, scale, acc);
}

// ---------------------------------------------------------------------------
// EDGE GEMM: for odd N / odd K / non-transposed B.
// Preconditions: M%128==0 (no row bounds).  N,K arbitrary.
// ---------------------------------------------------------------------------
template<int ACT, bool TRANS_B, bool HAS_BIAS>
__global__ void __launch_bounds__(256)
gemm_edge(const float* __restrict__ A, int lda, long long sA,
          const float* __restrict__ B, int ldb, long long sB,
          float* __restrict__ C, int ldc, long long sC,
          const float* __restrict__ bias,
          int N, int K, float scale)
{
    __shared__ __bf16 As[BM][BK + APAD];
    __shared__ __bf16 Bs[BN][BK + APAD];

    const int z = blockIdx.z;
    A += (long long)z * sA;
    B += (long long)z * sB;
    C += (long long)z * sC;

    const int m0 = blockIdx.y * BM;
    const int n0 = blockIdx.x * BN;
    const int tid  = threadIdx.x;
    const int lane = tid & 31;
    const int wave = tid >> 5;
    const int wm = (wave & 3) << 5;
    const int wn = (wave >> 2) << 5;

    v8f acc[2][2] = {};

    const int kTiles = (K + BK - 1) / BK;
    for (int kt = 0; kt < kTiles; ++kt) {
        const int k0 = kt * BK;
        // stage A: rows always valid; clamp K, select zero
        #pragma unroll 4
        for (int i = tid; i < BM * BK; i += 256) {
            int r = i >> 5, c = i & 31;
            int gk = k0 + c;
            bool ok = (gk < K);
            int ck = min(gk, K - 1);
            float v = A[(m0 + r) * lda + ck];
            As[r][c] = (__bf16)(ok ? v : 0.0f);
        }
        // stage B: clamp N and K, select zero
        #pragma unroll 4
        for (int i = tid; i < BN * BK; i += 256) {
            int r = i >> 5, c = i & 31;
            int gn = n0 + r, gk = k0 + c;
            bool ok = (gn < N) && (gk < K);
            int cn = min(gn, N - 1), ck = min(gk, K - 1);
            float v = TRANS_B ? B[cn * ldb + ck] : B[ck * ldb + cn];
            Bs[r][c] = (__bf16)(ok ? v : 0.0f);
        }
        __syncthreads();
        wmma_step<BK + APAD>(As, Bs, lane, wm, wn, acc);
        __syncthreads();
    }

    const int nn = lane & 15;
    const int mb = (lane >> 4) * 8;
    #pragma unroll
    for (int tj = 0; tj < 2; ++tj) {
        const int gn = n0 + wn + tj * 16 + nn;
        if (gn < N) {
            const float bv = HAS_BIAS ? bias[gn] : 0.0f;
            #pragma unroll
            for (int ti = 0; ti < 2; ++ti) {
                const int gm0 = m0 + wm + ti * 16 + mb;
                float* cp = C + gm0 * ldc + gn;
                #pragma unroll
                for (int r = 0; r < 8; ++r) {
                    float v = acc[ti][tj][r] * scale + bv;
                    if (ACT == 1) v = silu_f(v);
                    else if (ACT == 2) v = softplus_f(v);
                    *cp = v;
                    cp += ldc;
                }
            }
        }
    }
}

// ---------------------------------------------------------------------------
// RevIN
// ---------------------------------------------------------------------------
__global__ void revin_kernel(const float* __restrict__ x,
                             const float* __restrict__ w,
                             const float* __restrict__ bia,
                             float* __restrict__ xn,
                             float* __restrict__ mean,
                             float* __restrict__ stdv,
                             int L, int C)
{
    const int b = blockIdx.x;
    const int c = threadIdx.x;
    const float* xp = x + (long long)b * L * C;
    float s = 0.f, s2 = 0.f;
    for (int l = 0; l < L; ++l) {
        float v = xp[l * C + c];
        s += v; s2 += v * v;
    }
    const float m   = s / (float)L;
    const float var = fmaxf(s2 / (float)L - m * m, 0.0f);
    const float sd  = sqrtf(var + 1e-5f);
    mean[b * C + c] = m;
    stdv[b * C + c] = sd;
    const float inv = 1.0f / sd;
    float* op = xn + (long long)b * L * C;
    for (int l = 0; l < L; ++l) {
        float v = xp[l * C + c];
        op[l * C + c] = (v - m) * inv * w[c] + bia[c];
    }
}

// ---------------------------------------------------------------------------
// head split/merge
// ---------------------------------------------------------------------------
__global__ void split_heads_kernel(const float* __restrict__ qkv,
                                   float* __restrict__ q,
                                   float* __restrict__ k,
                                   float* __restrict__ v)
{
    int idx = blockIdx.x * blockDim.x + threadIdx.x;   // < 2^21
    const int total = 32 * 4 * 512 * 32;
    if (idx >= total) return;
    int d = idx & 31;
    int r = idx >> 5;
    int s = r & 511; r >>= 9;
    int h = r & 3;
    int b = r >> 2;
    int src = (b * 512 + s) * 384 + h * 32 + d;
    q[idx] = qkv[src];
    k[idx] = qkv[src + 128];
    v[idx] = qkv[src + 256];
}

__global__ void merge_heads_kernel(const float* __restrict__ oh,
                                   float* __restrict__ h2)
{
    int idx = blockIdx.x * blockDim.x + threadIdx.x;
    const int total = 32 * 4 * 512 * 32;
    if (idx >= total) return;
    int d = idx & 31;
    int r = idx >> 5;
    int s = r & 511; r >>= 9;
    int h = r & 3;
    int b = r >> 2;
    h2[(b * 512 + s) * 128 + h * 32 + d] = oh[idx];
}

// ---------------------------------------------------------------------------
// wave-per-row softmax (wave32)
// ---------------------------------------------------------------------------
__global__ void __launch_bounds__(256)
softmax_kernel(float* __restrict__ x, int rows, int n)
{
    const int row  = blockIdx.x * 8 + (threadIdx.x >> 5);
    const int lane = threadIdx.x & 31;
    if (row >= rows) return;
    float* p = x + (long long)row * n;
    float mx = -1e30f;
    for (int i = lane; i < n; i += 32) mx = fmaxf(mx, p[i]);
    for (int o = 16; o > 0; o >>= 1) mx = fmaxf(mx, __shfl_xor(mx, o, 32));
    float sum = 0.f;
    for (int i = lane; i < n; i += 32) { float e = __expf(p[i] - mx); p[i] = e; sum += e; }
    for (int o = 16; o > 0; o >>= 1) sum += __shfl_xor(sum, o, 32);
    const float inv = 1.0f / sum;
    for (int i = lane; i < n; i += 32) p[i] *= inv;
}

// ---------------------------------------------------------------------------
// batched 512x512 transpose
// ---------------------------------------------------------------------------
__global__ void transpose512_kernel(const float* __restrict__ in,
                                    float* __restrict__ out)
{
    __shared__ float tile[32][33];
    const int b  = blockIdx.z;
    const int c0 = blockIdx.x * 32;
    const int r0 = blockIdx.y * 32;
    const float* ip = in  + (long long)b * 512 * 512;
    float*       op = out + (long long)b * 512 * 512;
    for (int i = threadIdx.y; i < 32; i += 8)
        tile[i][threadIdx.x] = ip[(r0 + i) * 512 + c0 + threadIdx.x];
    __syncthreads();
    for (int i = threadIdx.y; i < 32; i += 8)
        op[(c0 + i) * 512 + r0 + threadIdx.x] = tile[threadIdx.x][i];
}

// ---------------------------------------------------------------------------
// causal depthwise conv (d_conv=4) + SiLU
// ---------------------------------------------------------------------------
__global__ void conv_silu_kernel(const float* __restrict__ xi, int ldin,
                                 const float* __restrict__ cw,
                                 const float* __restrict__ cb,
                                 float* __restrict__ xc,
                                 int Bn, int L, int D)
{
    int idx = blockIdx.x * blockDim.x + threadIdx.x;
    int total = Bn * L * D;
    if (idx >= total) return;
    int d = idx % D;
    int bl = idx / D;
    int l = bl % L;
    float acc = cb[d];
    #pragma unroll
    for (int k = 0; k < 4; ++k) {
        int ll = l + k - 3;
        if (ll >= 0)
            acc += xi[(bl + ll - l) * ldin + d] * cw[d * 4 + k];
    }
    xc[idx] = silu_f(acc);
}

// ---------------------------------------------------------------------------
// selective scan
// ---------------------------------------------------------------------------
__global__ void scan_kernel(const float* __restrict__ dt,
                            const float* __restrict__ xc,
                            const float* __restrict__ xdbl, int ldx,
                            int offB, int offC,
                            const float* __restrict__ A_log,
                            float* __restrict__ y,
                            int Bn, int L, int D)
{
    int tid = blockIdx.x * blockDim.x + threadIdx.x;
    if (tid >= Bn * D) return;
    const int b = tid / D;
    const int d = tid % D;
    float h[16];
    float Av[16];
    #pragma unroll
    for (int s = 0; s < 16; ++s) { h[s] = 0.f; Av[s] = -__expf(A_log[d * 16 + s]); }
    for (int t = 0; t < L; ++t) {
        const int bl = b * L + t;
        const float dtv = dt[bl * D + d];
        const float xv  = xc[bl * D + d];
        const float* Bp = xdbl + bl * ldx + offB;
        const float* Cp = xdbl + bl * ldx + offC;
        float yv = 0.f;
        #pragma unroll
        for (int s = 0; s < 16; ++s) {
            const float dA = __expf(dtv * Av[s]);
            h[s] = dA * h[s] + dtv * Bp[s] * xv;
            yv  += h[s] * Cp[s];
        }
        y[bl * D + d] = yv;
    }
}

__global__ void yact_kernel(float* __restrict__ y,
                            const float* __restrict__ xc,
                            const float* __restrict__ Dv,
                            const float* __restrict__ xz, int ldz, int zoff,
                            int total, int D)
{
    int idx = blockIdx.x * blockDim.x + threadIdx.x;
    if (idx >= total) return;
    int d = idx % D;
    int bl = idx / D;
    float z = xz[bl * ldz + zoff + d];
    y[idx] = (y[idx] + xc[idx] * Dv[d]) * silu_f(z);
}

__global__ void sub_kernel(float* __restrict__ out,
                           const float* __restrict__ a,
                           const float* __restrict__ b, int n)
{
    int i = blockIdx.x * blockDim.x + threadIdx.x;
    if (i < n) out[i] = a[i] - b[i];
}

__global__ void add3_kernel(float* __restrict__ a,
                            const float* __restrict__ b,
                            const float* __restrict__ c, int n)
{
    int i = blockIdx.x * blockDim.x + threadIdx.x;
    if (i < n) a[i] = a[i] + b[i] + c[i];
}

__global__ void final_kernel(const float* __restrict__ proj,
                             const float* __restrict__ mean,
                             const float* __restrict__ stdv,
                             const float* __restrict__ w,
                             const float* __restrict__ bia,
                             float* __restrict__ out)
{
    int idx = blockIdx.x * blockDim.x + threadIdx.x;
    const int total = 32 * 96 * 512;
    if (idx >= total) return;
    int c = idx % 512;
    int r = idx / 512;
    int t = r % 96;
    int b = r / 96;
    float v = proj[(b * 512 + c) * 96 + t];
    v = (v - bia[c]) / w[c] * stdv[b * 512 + c] + mean[b * 512 + c];
    out[idx] = v;
}

// ---------------------------------------------------------------------------
// host side
// ---------------------------------------------------------------------------
template<int ACT, bool HB, bool HR>
static void fastg(hipStream_t st,
                  const float* A, int lda, long long sA,
                  const float* B, int ldb, long long sB,
                  float* C, int ldc, long long sC,
                  const float* bias,
                  const float* R, int ldr, long long sR,
                  int M, int N, int K, float scale, int batch)
{
    dim3 grid(N / BN, M / BM, batch);
    gemm_fast<ACT, HB, HR><<<grid, dim3(256), 0, st>>>(
        A, lda, sA, B, ldb, sB, C, ldc, sC, bias, R, ldr, sR, K, scale);
}

template<int ACT, bool TB, bool HB>
static void edgeg(hipStream_t st,
                  const float* A, int lda, long long sA,
                  const float* B, int ldb, long long sB,
                  float* C, int ldc, long long sC,
                  const float* bias,
                  int M, int N, int K, float scale, int batch)
{
    dim3 grid((N + BN - 1) / BN, M / BM, batch);
    gemm_edge<ACT, TB, HB><<<grid, dim3(256), 0, st>>>(
        A, lda, sA, B, ldb, sB, C, ldc, sC, bias, N, K, scale);
}

static inline dim3 ew_grid(long long n) { return dim3((unsigned)((n + 255) / 256)); }

extern "C" void kernel_launch(void* const* d_in, const int* in_sizes, int n_in,
                              void* d_out, int out_size, void* d_ws, size_t ws_size,
                              hipStream_t stream)
{
    (void)in_sizes; (void)n_in; (void)out_size; (void)ws_size;

    const float* x_enc   = (const float*)d_in[0];
    const float* W1      = (const float*)d_in[1];
    const float* b1      = (const float*)d_in[2];
    const float* in_proj = (const float*)d_in[3];
    const float* in_b    = (const float*)d_in[4];
    const float* Wo      = (const float*)d_in[5];
    const float* bo      = (const float*)d_in[6];
    const float* Ws      = (const float*)d_in[7];
    const float* bs      = (const float*)d_in[8];
    const float* Wt      = (const float*)d_in[9];
    const float* bt      = (const float*)d_in[10];
    const float* emb_W   = (const float*)d_in[11];
    const float* emb_b   = (const float*)d_in[12];
    const float* emb_s_W = (const float*)d_in[13];
    const float* emb_s_b = (const float*)d_in[14];
    const float* emb_t_W = (const float*)d_in[15];
    const float* emb_t_b = (const float*)d_in[16];
    const float* Tprj_W  = (const float*)d_in[17];
    const float* Tprj_b  = (const float*)d_in[18];
    const float* Tback_W = (const float*)d_in[19];
    const float* Tback_b = (const float*)d_in[20];
    const float* proj_W  = (const float*)d_in[21];
    const float* proj_b  = (const float*)d_in[22];
    const float* revin_w = (const float*)d_in[23];
    const float* revin_b = (const float*)d_in[24];
    const float* mS_in_W     = (const float*)d_in[25];
    const float* mS_conv_w   = (const float*)d_in[26];
    const float* mS_conv_b   = (const float*)d_in[27];
    const float* mS_x_proj_W = (const float*)d_in[28];
    const float* mS_dt_W     = (const float*)d_in[29];
    const float* mS_dt_b     = (const float*)d_in[30];
    const float* mS_A_log    = (const float*)d_in[31];
    const float* mS_D        = (const float*)d_in[32];
    const float* mS_out_W    = (const float*)d_in[33];
    const float* mT_in_W     = (const float*)d_in[34];
    const float* mT_conv_w   = (const float*)d_in[35];
    const float* mT_conv_b   = (const float*)d_in[36];
    const float* mT_x_proj_W = (const float*)d_in[37];
    const float* mT_dt_W     = (const float*)d_in[38];
    const float* mT_dt_b     = (const float*)d_in[39];
    const float* mT_A_log    = (const float*)d_in[40];
    const float* mT_D        = (const float*)d_in[41];
    const float* mT_out_W    = (const float*)d_in[42];

    float* ws  = (float*)d_ws;
    float* out = (float*)d_out;

    const long long MF = 1048576LL;
    float* slabA = ws + 0 * MF;       // 32M: att | mS xz | mT temps
    float* slabB = ws + 32 * MF;      // 16M: xn+h+qkv | mS xc
    float* slabC = ws + 48 * MF;      // 16M: h2,trend,resid,senc | mS dt
    float* slabD = ws + 64 * MF;      // 16M: q,k,v,oh | tenc,sT | mS y
    float* slabE = ws + 80 * MF;      // 8M : tT | seasonal
    float* x_e   = ws + 88 * MF;      // 8M
    float* slabG = ws + 96 * MF;      // 8M : s_e | trend2
    float* t_e   = ws + 104 * MF;     // 8M
    float* xdbl  = ws + 112 * MF;     // 1M
    float* x_t   = ws + 113 * MF;     // 1M
    float* xt2   = ws + 114 * MF;     // 1M
    float* projo = ws + 115 * MF;     // 1.5M
    float* meanb = ws + 117 * MF;
    float* stdb  = meanb + 16384;

    const int M = 32 * 512;           // 16384
    const float inv_sqrt_hd = 0.17677669529663687f;

    float* xn  = slabB;
    float* h   = slabB + 8 * MF;
    float* qkv = slabB + 10 * MF;
    revin_kernel<<<dim3(32), dim3(512), 0, stream>>>(x_enc, revin_w, revin_b,
                                                     xn, meanb, stdb, 512, 512);

    fastg<0, true, false>(stream, xn, 512, 0, W1, 512, 0, h, 128, 0,
                          b1, nullptr, 0, 0, M, 128, 512, 1.0f, 1);

    fastg<0, true, false>(stream, h, 128, 0, in_proj, 128, 0, qkv, 384, 0,
                          in_b, nullptr, 0, 0, M, 384, 128, 1.0f, 1);

    float* q  = slabD;
    float* k  = slabD + 2 * MF;
    float* v  = slabD + 4 * MF;
    float* oh = slabD + 6 * MF;
    {
        long long n = 32LL * 4 * 512 * 32;
        split_heads_kernel<<<ew_grid(n), dim3(256), 0, stream>>>(qkv, q, k, v);
    }

    float* att = slabA;
    fastg<0, false, false>(stream, q, 32, 512 * 32, k, 32, 512 * 32,
                           att, 512, 512LL * 512, nullptr, nullptr, 0, 0,
                           512, 512, 32, inv_sqrt_hd, 128);
    softmax_kernel<<<dim3(128 * 512 / 8), dim3(256), 0, stream>>>(att, 128 * 512, 512);

    edgeg<0, false, false>(stream, att, 512, 512LL * 512, v, 32, 512 * 32,
                           oh, 32, 512 * 32, nullptr,
                           512, 32, 512, 1.0f, 128);

    float* h2    = slabC;
    float* trend = slabC + 2 * MF;
    float* resid = slabC + 4 * MF;
    float* senc  = slabC + 6 * MF;
    {
        long long n = 32LL * 4 * 512 * 32;
        merge_heads_kernel<<<ew_grid(n), dim3(256), 0, stream>>>(oh, h2);
    }

    fastg<1, true, false>(stream, h2, 128, 0, Wo, 128, 0, trend, 128, 0,
                          bo, nullptr, 0, 0, M, 128, 128, 1.0f, 1);

    sub_kernel<<<ew_grid((long long)M * 128), dim3(256), 0, stream>>>(
        resid, h, trend, M * 128);

    float* tenc = slabD;
    float* sT   = slabD + 8 * MF;
    fastg<0, true, false>(stream, trend, 128, 0, Ws, 128, 0, senc, 512, 0,
                          bs, nullptr, 0, 0, M, 512, 128, 1.0f, 1);
    fastg<0, true, false>(stream, resid, 128, 0, Wt, 128, 0, tenc, 512, 0,
                          bt, nullptr, 0, 0, M, 512, 128, 1.0f, 1);

    float* tT = slabE;
    transpose512_kernel<<<dim3(16, 16, 32), dim3(32, 8), 0, stream>>>(senc, sT);
    transpose512_kernel<<<dim3(16, 16, 32), dim3(32, 8), 0, stream>>>(tenc, tT);

    float* s_e = slabG;
    fastg<0, true, false>(stream, sT, 512, 0, emb_W, 512, 0, x_e, 512, 0,
                          emb_b, nullptr, 0, 0, M, 512, 512, 1.0f, 1);
    fastg<0, true, false>(stream, sT, 512, 0, emb_s_W, 512, 0, s_e, 512, 0,
                          emb_s_b, nullptr, 0, 0, M, 512, 512, 1.0f, 1);
    fastg<0, true, false>(stream, tT, 512, 0, emb_t_W, 512, 0, t_e, 512, 0,
                          emb_t_b, nullptr, 0, 0, M, 512, 512, 1.0f, 1);

    // ================= Mamba mS (d_inner=1024) =================
    float* xz = slabA;
    float* xc = slabB;
    float* dt = slabC;
    float* y  = slabD;
    fastg<0, false, false>(stream, s_e, 512, 0, mS_in_W, 512, 0, xz, 2048, 0,
                           nullptr, nullptr, 0, 0, M, 2048, 512, 1.0f, 1);
    conv_silu_kernel<<<ew_grid((long long)M * 1024), dim3(256), 0, stream>>>(
        xz, 2048, mS_conv_w, mS_conv_b, xc, 32, 512, 1024);
    fastg<0, false, false>(stream, xc, 1024, 0, mS_x_proj_W, 1024, 0, xdbl, 64, 0,
                           nullptr, nullptr, 0, 0, M, 64, 1024, 1.0f, 1);
    fastg<2, true, false>(stream, xdbl, 64, 0, mS_dt_W, 32, 0, dt, 1024, 0,
                          mS_dt_b, nullptr, 0, 0, M, 1024, 32, 1.0f, 1);
    scan_kernel<<<ew_grid(32LL * 1024), dim3(256), 0, stream>>>(
        dt, xc, xdbl, 64, 32, 48, mS_A_log, y, 32, 512, 1024);
    yact_kernel<<<ew_grid((long long)M * 1024), dim3(256), 0, stream>>>(
        y, xc, mS_D, xz, 2048, 1024, M * 1024, 1024);
    float* seasonal = slabE;
    fastg<0, false, false>(stream, y, 1024, 0, mS_out_W, 1024, 0, seasonal, 512, 0,
                           nullptr, nullptr, 0, 0, M, 512, 1024, 1.0f, 1);

    // ================= trend branch: Tprj + Mamba mT =================
    fastg<0, true, false>(stream, t_e, 512, 0, Tprj_W, 512, 0, x_t, 64, 0,
                          Tprj_b, nullptr, 0, 0, M, 64, 512, 1.0f, 1);
    float* xz2   = slabA;
    float* xc2   = slabA + 4 * MF;
    float* xdbl2 = slabA + 6 * MF;
    float* dt2   = slabA + 8 * MF;
    float* y2    = slabA + 10 * MF;
    fastg<0, false, false>(stream, x_t, 64, 0, mT_in_W, 64, 0, xz2, 256, 0,
                           nullptr, nullptr, 0, 0, M, 256, 64, 1.0f, 1);
    conv_silu_kernel<<<ew_grid((long long)M * 128), dim3(256), 0, stream>>>(
        xz2, 256, mT_conv_w, mT_conv_b, xc2, 32, 512, 128);
    edgeg<0, true, false>(stream, xc2, 128, 0, mT_x_proj_W, 128, 0, xdbl2, 36, 0,
                          nullptr, M, 36, 128, 1.0f, 1);
    edgeg<2, true, true>(stream, xdbl2, 36, 0, mT_dt_W, 4, 0, dt2, 128, 0,
                         mT_dt_b, M, 128, 4, 1.0f, 1);
    scan_kernel<<<ew_grid(32LL * 128), dim3(256), 0, stream>>>(
        dt2, xc2, xdbl2, 36, 4, 20, mT_A_log, y2, 32, 512, 128);
    yact_kernel<<<ew_grid((long long)M * 128), dim3(256), 0, stream>>>(
        y2, xc2, mT_D, xz2, 256, 128, M * 128, 128);
    fastg<0, false, true>(stream, y2, 128, 0, mT_out_W, 128, 0, xt2, 64, 0,
                          nullptr, x_t, 64, 0, M, 64, 128, 1.0f, 1);
    float* trend2 = slabG;
    fastg<0, true, false>(stream, xt2, 64, 0, Tback_W, 64, 0, trend2, 512, 0,
                          Tback_b, nullptr, 0, 0, M, 512, 64, 1.0f, 1);

    add3_kernel<<<ew_grid((long long)M * 512), dim3(256), 0, stream>>>(
        seasonal, trend2, x_e, M * 512);

    edgeg<0, true, true>(stream, seasonal, 512, 0, proj_W, 512, 0, projo, 96, 0,
                         proj_b, M, 96, 512, 1.0f, 1);
    final_kernel<<<ew_grid(32LL * 96 * 512), dim3(256), 0, stream>>>(
        projo, meanb, stdb, revin_w, revin_b, out);
}
